// LayerStacks_20074677141704
// MI455X (gfx1250) — matrix operand order repaired and checked by reading, not compiled, not run on hardware
//
#include <hip/hip_runtime.h>

typedef __attribute__((ext_vector_type(16))) _Float16 v16h;
typedef __attribute__((ext_vector_type(8)))  float    v8f;

#define L1PA 65
#define NB   6          // COUNT buckets
#define N1   48         // 6*8 layer-1 neurons (all buckets)
#define N2   192        // 6*32 layer-2 neurons (all buckets)
#define NFRAG 9         // 3 M-tiles x 3 K-chunks of the layer-1 GEMM

__device__ __forceinline__ float q_round(float w, float s) { return rintf(w * s) / s; }

__global__ __launch_bounds__(256) void layerstacks_kernel(
    const float* __restrict__ x,  const int* __restrict__ ply,
    const float* __restrict__ w1, const float* __restrict__ b1,
    const float* __restrict__ w2, const float* __restrict__ b2,
    const float* __restrict__ w3, const float* __restrict__ b3,
    float* __restrict__ out, int n)
{
    // w1 pre-quantized, converted to f16 and pre-swizzled into the WMMA A-matrix
    // register layout, one 16x32 fragment per (tile, chunk): 9 x 512 f16 = 9 KB.
    // Per-wave fragment fetch is then two ds_load_b128 per lane, zero VALU.
    __shared__ alignas(32) _Float16 s_w1h[NFRAG * 512];
    __shared__ float s_b1[N1];
    __shared__ float s_w2[N2 * 8];
    __shared__ float s_b2[N2];
    __shared__ float s_w3[N2];
    __shared__ float s_b3[NB];

    const int tid = threadIdx.x;

    // ---- one-time per-block setup: quantize weights, pack A fragments ----
    for (int i = tid; i < NFRAG * 512; i += 256) {
        const int f  = i >> 9;           // fragment index = c*3 + t
        const int r  = i & 511;
        const int ln = r >> 4;           // destination lane within the wave
        const int e  = r & 15;           // f16 element within that lane's v16h
        const int c  = f / 3;
        const int t  = f - 3 * c;
        const int mm = ln & 15;
        const int hh = ln >> 4;
        const int j  = e >> 1;
        const int h  = e & 1;
        // A-matrix 16x32 f16 layout: lanes 0-15 hold K=0-7,16-23; lanes 16-31 K=8-15,24-31
        const int K   = 32 * c + ((j >> 2) << 4) + 8 * hh + 2 * (j & 3) + h;
        const int row = 16 * t + mm;
        const float v = (K < L1PA) ? q_round(w1[row * L1PA + K], 64.f) : 0.f;
        s_w1h[i] = (_Float16)v;
    }
    for (int i = tid; i < N1;     i += 256) s_b1[i] = q_round(b1[i], 64.f * 127.f);
    for (int i = tid; i < N2 * 8; i += 256) s_w2[i] = q_round(w2[i], 64.f);
    for (int i = tid; i < N2;     i += 256) s_b2[i] = q_round(b2[i], 64.f * 127.f);
    const float oscale = 32.f * 16.f / 127.f;
    for (int i = tid; i < N2;     i += 256) s_w3[i] = q_round(w3[i], oscale);
    for (int i = tid; i < NB;     i += 256) s_b3[i] = q_round(b3[i], 32.f * 16.f);
    __syncthreads();

    const int wave = tid >> 5;
    const int lane = tid & 31;
    const int m    = lane & 15;      // sample-in-tile (= N column)
    const int hi   = lane >> 4;      // lane half: K-half for B, M-half for C/D

    const int sBase = (blockIdx.x * 8 + wave) * 16;
    const int s     = sBase + m;
    const int sc    = (s < n) ? s : (n - 1);   // clamp: EXEC stays full for WMMA

    int b = ply[sc] / 10;                      // bucket = ply // BUCKET
    b = b < 0 ? 0 : (b > 5 ? 5 : b);

    // C/D f32 16x16 layout: VGPR r, lanes 0-15 -> M=r, lanes 16-31 -> M=r+8
    v8f acc0, acc1, acc2;
    #pragma unroll
    for (int r = 0; r < 8; r++) {
        acc0[r] = s_b1[ 0 + r + 8 * hi];
        acc1[r] = s_b1[16 + r + 8 * hi];
        acc2[r] = s_b1[32 + r + 8 * hi];
    }

    // Layer 1 as W1q(48x65) @ X^T(65x16): 3 M-tiles x 3 K-chunks (65 -> pad 96)
    const float* xrow = x + (long)sc * L1PA;
    const v16h*  afrags = reinterpret_cast<const v16h*>(s_w1h);
    #pragma unroll
    for (int c = 0; c < 3; c++) {
        // B fragment (32x16 f16): lanes 0-15 hold K=0-7,16-23; lanes 16-31 K=8-15,24-31; column = lane%16.
        // x is streamed once -> non-temporal loads (TH=NT), converted to f16.
        v16h bf;
        const int kb = 32 * c + 8 * hi;
        #pragma unroll
        for (int j = 0; j < 16; j++) {
            const int K = kb + j + ((j >> 3) << 3);
            bf[j] = (_Float16)((K < L1PA) ? __builtin_nontemporal_load(&xrow[K]) : 0.f);
        }
        #pragma unroll
        for (int t = 0; t < 3; t++) {
            const v16h af = afrags[(c * 3 + t) * 32 + lane];
            v8f& acc = (t == 0) ? acc0 : ((t == 1) ? acc1 : acc2);
            acc = __builtin_amdgcn_wmma_f32_16x16x32_f16(
                false, af, false, bf, (short)0, acc, false, false);
        }
    }

    // Bucket select: neurons 8b..8b+7 live in tile b>>1, lane half (b&1).
    // One shfl_xor(16) gives both lanes of the pair the full 8-vector.
    const int  tsel    = b >> 1;
    const bool useSelf = ((b & 1) == hi);
    float l1v[8];
    #pragma unroll
    for (int r = 0; r < 8; r++) {
        float v = (tsel == 0) ? acc0[r] : ((tsel == 1) ? acc1[r] : acc2[r]);
        float o = __shfl_xor(v, 16, 32);
        float sel = useSelf ? v : o;
        sel = fminf(fmaxf(sel, 0.f), 1.f);
        l1v[r] = floorf(sel * 127.f) / 127.f;
    }

    // Layers 2+3 (bucket-selected, split across the lane pair): 16 of 32 outputs each
    const int obase = 32 * b + 16 * hi;
    float p = 0.f;
    #pragma unroll
    for (int i = 0; i < 16; i++) {
        const int o = obase + i;
        float a = s_b2[o];
        #pragma unroll
        for (int k = 0; k < 8; k++) a += l1v[k] * s_w2[o * 8 + k];
        a = fminf(fmaxf(a, 0.f), 1.f);
        a = floorf(a * 127.f) / 127.f;
        p += a * s_w3[o];
    }
    const float tot = p + __shfl_xor(p, 16, 32);
    if (hi == 0 && s < n) {
        const float r3 = floorf((tot + s_b3[b]) * 32.f) / 32.f;
        __builtin_nontemporal_store(r3, &out[s]);
    }
}

extern "C" void kernel_launch(void* const* d_in, const int* in_sizes, int n_in,
                              void* d_out, int out_size, void* d_ws, size_t ws_size,
                              hipStream_t stream)
{
    const float* x   = (const float*)d_in[0];
    const int*   ply = (const int*)  d_in[1];
    const float* w1  = (const float*)d_in[2];
    const float* b1  = (const float*)d_in[3];
    const float* w2  = (const float*)d_in[4];
    const float* b2  = (const float*)d_in[5];
    const float* w3  = (const float*)d_in[6];
    const float* b3  = (const float*)d_in[7];
    float* out = (float*)d_out;

    const int n = in_sizes[1];                 // batch size B (ply count)
    const int blocks = (n + 127) / 128;        // 8 waves/block x 16 samples/wave
    layerstacks_kernel<<<blocks, 256, 0, stream>>>(x, ply, w1, b1, w2, b2, w3, b3, out, n);
}
